// mLSTMCell_41042707480756
// MI455X (gfx1250) — matrix operand order repaired and checked by reading, not compiled, not run on hardware
//
#include <hip/hip_runtime.h>

typedef __attribute__((ext_vector_type(16))) __bf16 v16bf;
typedef __attribute__((ext_vector_type(8)))  __bf16 v8bf;
typedef __attribute__((ext_vector_type(8)))  float  v8f;

#define B_SZ   2048
#define IN_DIM 4096
#define HID    2048
#define NH     64
#define DH     32
#define PROJ   8320

// GEMM tiling: 128x128 block, BK=32, 256 threads = 8 waves (2 x 4 wave grid),
// each wave owns a 64x32 sub-tile = 8 accumulators of 16x16 f32.
#define BM 128
#define BN 128
#define BK 32
#define LDS_K 40   // 32 bf16 + 8 pad (80 bytes/row) to spread LDS banks

static __device__ __forceinline__ v16bf make_v16(v8bf lo, v8bf hi) {
    v16bf r;
#pragma unroll
    for (int e = 0; e < 8; ++e) { r[e] = lo[e]; r[e + 8] = hi[e]; }
    return r;
}

__global__ __launch_bounds__(256)
void mlstm_gemm_bf16wmma(const float* __restrict__ X,
                         const float* __restrict__ W,
                         const float* __restrict__ bias,
                         float* __restrict__ P)
{
    __shared__ __bf16 As[BM][LDS_K];   // x tile:  [row][k]
    __shared__ __bf16 Bs[BN][LDS_K];   // W tile, transposed: [col][k]

    const int tid  = threadIdx.x;
    const int lane = tid & 31;
    const int wave = tid >> 5;
    const int l16  = lane & 15;
    const int half = lane >> 4;        // 0: lanes 0-15, 1: lanes 16-31
    const int wm   = wave & 1;         // wave row  (2 x 64 rows)
    const int wn   = wave >> 1;        // wave col  (4 x 32 cols)

    const int rowBlk = blockIdx.y * BM;
    const int colBlk = blockIdx.x * BN;

    // A fill coordinates: each thread converts 16 contiguous f32 of x -> 16 bf16
    const int arow  = tid >> 1;              // 0..127
    const int akseg = (tid & 1) * 16;        // 0 or 16
    // B fill coordinates: each thread reads a 16-deep K column of W (coalesced
    // across the 128 threads sharing each row) and stores it contiguously.
    const int bcol = tid & 127;              // 0..127
    const int bks  = (tid >> 7) * 16;        // 0 or 16

    v8f acc[4][2];
#pragma unroll
    for (int mt = 0; mt < 4; ++mt)
#pragma unroll
        for (int nt = 0; nt < 2; ++nt)
            acc[mt][nt] = (v8f){0.f, 0.f, 0.f, 0.f, 0.f, 0.f, 0.f, 0.f};

    for (int k0 = 0; k0 < IN_DIM; k0 += BK) {
        // ---- stage tiles into LDS (f32 -> bf16 on the fly; x+W live in L2) ----
        const float4* asrc =
            (const float4*)(X + (size_t)(rowBlk + arow) * IN_DIM + k0 + akseg);
        float4 f0 = asrc[0], f1 = asrc[1], f2 = asrc[2], f3 = asrc[3];

        const float* wsrc = W + (size_t)(k0 + bks) * PROJ + colBlk + bcol;
        float wv[16];
#pragma unroll
        for (int j = 0; j < 16; ++j) wv[j] = wsrc[(size_t)j * PROJ];

        if (k0 + BK < IN_DIM) {   // pull next K-slice toward L2 (global_prefetch_b8)
            __builtin_prefetch(asrc + 4 * (BK / 16), 0, 1);
            __builtin_prefetch(wsrc + (size_t)BK * PROJ, 0, 1);
        }

        v8bf a0, a1, b0, b1;
        a0[0]=(__bf16)f0.x; a0[1]=(__bf16)f0.y; a0[2]=(__bf16)f0.z; a0[3]=(__bf16)f0.w;
        a0[4]=(__bf16)f1.x; a0[5]=(__bf16)f1.y; a0[6]=(__bf16)f1.z; a0[7]=(__bf16)f1.w;
        a1[0]=(__bf16)f2.x; a1[1]=(__bf16)f2.y; a1[2]=(__bf16)f2.z; a1[3]=(__bf16)f2.w;
        a1[4]=(__bf16)f3.x; a1[5]=(__bf16)f3.y; a1[6]=(__bf16)f3.z; a1[7]=(__bf16)f3.w;
#pragma unroll
        for (int j = 0; j < 8; ++j) { b0[j] = (__bf16)wv[j]; b1[j] = (__bf16)wv[j + 8]; }

        __syncthreads();
        *(v8bf*)&As[arow][akseg]     = a0;
        *(v8bf*)&As[arow][akseg + 8] = a1;
        *(v8bf*)&Bs[bcol][bks]       = b0;
        *(v8bf*)&Bs[bcol][bks + 8]   = b1;
        __syncthreads();

        // ---- fragments (ISA 16-bit A/B VGPR layouts) ----
        v16bf afrag[4], bfrag[2];
#pragma unroll
        for (int mt = 0; mt < 4; ++mt) {
            const __bf16* pa = &As[wm * 64 + mt * 16 + l16][half * 8];
            // VGPR0-3: K = half*8 + 0..7 ; VGPR4-7: K = half*8 + 16..23
            afrag[mt] = make_v16(*(const v8bf*)pa, *(const v8bf*)(pa + 16));
        }
#pragma unroll
        for (int nt = 0; nt < 2; ++nt) {
            const __bf16* pb = &Bs[wn * 32 + nt * 16 + l16][half * 8];
            // VGPR0-3: K = half*8 + 0..7 ; VGPR4-7: K = half*8 + 16..23
            bfrag[nt] = make_v16(*(const v8bf*)pb, *(const v8bf*)(pb + 16));
        }

#pragma unroll
        for (int mt = 0; mt < 4; ++mt)
#pragma unroll
            for (int nt = 0; nt < 2; ++nt)
                acc[mt][nt] = __builtin_amdgcn_wmma_f32_16x16x32_bf16(
                    false, afrag[mt], false, bfrag[nt],
                    (short)0, acc[mt][nt], false, false);
    }

    // ---- epilogue: D layout is M = vgpr + 8*half, N = lane%16 ----
#pragma unroll
    for (int mt = 0; mt < 4; ++mt) {
#pragma unroll
        for (int nt = 0; nt < 2; ++nt) {
            const int col = colBlk + wn * 32 + nt * 16 + l16;
            const float bv = bias[col];
            const int rbase = rowBlk + wm * 64 + mt * 16 + half * 8;
#pragma unroll
            for (int v = 0; v < 8; ++v)
                P[(size_t)(rbase + v) * PROJ + col] = acc[mt][nt][v] + bv;
        }
    }
}

// One wave per (batch, head); lane = column j of the 32x32 head state.
// All C traffic (the 1 GB that dominates) is fully coalesced.
__global__ __launch_bounds__(256)
void mlstm_cell_update(const float* __restrict__ P,
                       const float* __restrict__ C_prev,
                       const float* __restrict__ n_prev,
                       const float* __restrict__ m_prev,
                       float* __restrict__ h_out,
                       float* __restrict__ C_out,
                       float* __restrict__ n_out,
                       float* __restrict__ m_out)
{
    const int wave = threadIdx.x >> 5;
    const int lane = threadIdx.x & 31;
    const int bh   = blockIdx.x * 8 + wave;   // 0 .. B*NH-1
    const int b    = bh >> 6;
    const int h    = bh & 63;

    const float* prow = P + (size_t)b * PROJ;
    const float q  = prow[0 * HID + h * DH + lane];
    const float k  = prow[1 * HID + h * DH + lane];
    const float v  = prow[2 * HID + h * DH + lane];
    const float ig = prow[3 * HID + h];
    const float fg = prow[3 * HID + NH + h];
    float og = prow[3 * HID + 2 * NH + h * DH + lane];
    og = 1.0f / (1.0f + __expf(-og));

    const float mp = m_prev[bh];
    const float mt = fmaxf(fg + mp, ig);
    const float ip = __expf(ig - mt);
    const float fp = __expf(fg + mp - mt);

    const float nt = fp * n_prev[(size_t)bh * DH + lane] + ip * k;
    n_out[(size_t)bh * DH + lane] = nt;

    float d = nt * q;
#pragma unroll
    for (int off = 16; off; off >>= 1) d += __shfl_xor(d, off, 32);
    const float den = fabsf(d) + 1e-6f;

    const float* Cp = C_prev + (size_t)bh * (DH * DH);
    float*       Co = C_out  + (size_t)bh * (DH * DH);
    float mynum = 0.0f;
    for (int i = 0; i < DH; ++i) {
        const float vi = __shfl(v, i, 32);
        const float c  = fp * Cp[i * DH + lane] + ip * vi * k;   // coalesced 128B
        Co[i * DH + lane] = c;                                   // coalesced 128B
        float r = c * q;
#pragma unroll
        for (int off = 16; off; off >>= 1) r += __shfl_xor(r, off, 32);
        if (lane == i) mynum = r;   // lane i keeps num_i
    }

    h_out[(size_t)b * HID + h * DH + lane] = og * mynum / den;
    if (lane == 0) m_out[bh] = mt;
}

extern "C" void kernel_launch(void* const* d_in, const int* in_sizes, int n_in,
                              void* d_out, int out_size, void* d_ws, size_t ws_size,
                              hipStream_t stream)
{
    const float* x  = (const float*)d_in[0];
    const float* Cp = (const float*)d_in[1];
    const float* np = (const float*)d_in[2];
    const float* mp = (const float*)d_in[3];
    const float* W  = (const float*)d_in[4];
    const float* bv = (const float*)d_in[5];
    float* out = (float*)d_out;
    float* P   = (float*)d_ws;          // 2048 x 8320 f32 projections (~65 MB)

    dim3 g1(PROJ / BN, B_SZ / BM);      // 65 x 16 workgroups
    mlstm_gemm_bf16wmma<<<g1, 256, 0, stream>>>(x, W, bv, P);

    const size_t OFF_H = 0;
    const size_t OFF_C = (size_t)B_SZ * HID;
    const size_t OFF_N = OFF_C + (size_t)B_SZ * NH * DH * DH;
    const size_t OFF_M = OFF_N + (size_t)B_SZ * NH * DH;
    mlstm_cell_update<<<(B_SZ * NH) / 8, 256, 0, stream>>>(
        P, Cp, np, mp, out + OFF_H, out + OFF_C, out + OFF_N, out + OFF_M);
}
